// BidirectionalSimplifiedSSM_60567628808650
// MI455X (gfx1250) — compile-verified, matrix operand order checked by reading, not run on hardware
//
#include <hip/hip_runtime.h>
#include <hip/hip_bf16.h>

// ---------- types ----------
typedef __attribute__((ext_vector_type(16))) __bf16 bf16x16;
typedef __attribute__((ext_vector_type(8)))  float  f32x8;
typedef __attribute__((ext_vector_type(4)))  unsigned int u32x4;
typedef __attribute__((ext_vector_type(4)))  int          i32x4;
typedef __attribute__((ext_vector_type(8)))  int          i32x8;

union FragU {
    bf16x16 v;
    unsigned int u[8];
};

#if defined(__has_builtin)
#if __has_builtin(__builtin_amdgcn_tensor_load_to_lds) && __has_builtin(__builtin_amdgcn_s_wait_tensorcnt)
#define HAVE_TDM 1
#endif
#endif
#ifndef HAVE_TDM
#define HAVE_TDM 0
#endif

__device__ __forceinline__ unsigned short f2bf(float f) {
    union { float f; unsigned int u; } c;
    c.f = f;
    unsigned int u = c.u;
    unsigned int r = u + 0x7FFFu + ((u >> 16) & 1u);   // round-to-nearest-even
    return (unsigned short)(r >> 16);
}

__device__ __forceinline__ float sigmoid_f(float x) {
    return 1.0f / (1.0f + __expf(-x));
}

// ---------- TDM helper: 2D tile (nrows x row_elems f32) global -> LDS ----------
// Descriptor layout per CDNA5 ISA (D# groups 0/1; groups 2/3 zero for 2D).
#if HAVE_TDM
__device__ __forceinline__ void tdm_load_2d_f32(const float* gptr, unsigned lds_byte_off,
                                                unsigned row_elems, unsigned nrows,
                                                unsigned row_stride_elems) {
    unsigned long long ga = (unsigned long long)(uintptr_t)gptr;
    u32x4 g0;
    g0[0] = 1u;                                            // count=1 (valid), is_restore=0
    g0[1] = lds_byte_off;                                  // lds_addr
    g0[2] = (unsigned)(ga & 0xFFFFFFFFull);                // global_addr[31:0]
    g0[3] = (unsigned)((ga >> 32) & 0x01FFFFFFull) | (2u << 30);  // addr[56:32] | type=2

    const unsigned td0 = row_elems;                        // tensor_dim0 (OOB bound)
    const unsigned td1 = 1u << 20;                         // tensor_dim1 (generous bound)
    const unsigned s0  = row_stride_elems;                 // tensor_dim0_stride
    i32x8 g1;
    g1[0] = (int)(2u << 16);                               // workgroup_mask=0, data_size=2 (4B)
    g1[1] = (int)((td0 & 0xFFFFu) << 16);                  // [31:16]=tensor_dim0[15:0]
    g1[2] = (int)(((td0 >> 16) & 0xFFFFu) | ((td1 & 0xFFFFu) << 16));  // dim0 hi | dim1 lo
    g1[3] = (int)(((td1 >> 16) & 0xFFFFu) | ((row_elems & 0xFFFFu) << 16)); // dim1 hi | tile_dim0
    g1[4] = (int)(nrows & 0xFFFFu);                        // tile_dim1, tile_dim2=0
    g1[5] = (int)s0;                                       // stride0[31:0]
    g1[6] = 0;                                             // stride0[47:32]=0, stride1 lo=0
    g1[7] = 0;                                             // stride1 hi=0
    i32x4 gz = {};
#if __clang_major__ >= 23
    i32x8 gz8 = {};
    __builtin_amdgcn_tensor_load_to_lds(g0, g1, gz, gz, gz8, 0);
#else
    __builtin_amdgcn_tensor_load_to_lds(g0, g1, gz, gz, 0);
#endif
}
#endif

// ---------- convert kernels ----------
__global__ __launch_bounds__(256) void f32_to_bf16_copy(const float* __restrict__ src,
                                                        unsigned short* __restrict__ dst,
                                                        size_t n) {
    size_t i = (size_t)blockIdx.x * blockDim.x + threadIdx.x;
    if (i < n) dst[i] = f2bf(src[i]);
}

// src: [K,N] f32 row-major  ->  dst: [N,K] bf16 row-major (i.e. transposed)
__global__ __launch_bounds__(256) void transpose_f32_to_bf16(const float* __restrict__ src,
                                                             unsigned short* __restrict__ dst,
                                                             int K, int N) {
    size_t i = (size_t)blockIdx.x * blockDim.x + threadIdx.x;
    size_t total = (size_t)K * N;
    if (i >= total) return;
    size_t n = i / K;
    size_t k = i % K;
    dst[i] = f2bf(src[k * (size_t)N + n]);
}

// ---------- WMMA helpers ----------
__device__ __forceinline__ f32x8 wmma_bf16(const FragU& a, const FragU& b, f32x8 c) {
    return __builtin_amdgcn_wmma_f32_16x16x32_bf16(false, a.v, false, b.v,
                                                   (short)0, c, false, false);
}

__device__ __forceinline__ void store_tile(const f32x8& acc, int mbase, int nbase,
                                           int hi, int sub,
                                           float* __restrict__ C,
                                           unsigned short* __restrict__ Cbf,
                                           const float* __restrict__ resid, int N) {
#pragma unroll
    for (int r = 0; r < 8; ++r) {
        const int m = mbase + r + 8 * hi;
        const size_t idx = (size_t)m * N + nbase + sub;
        float v = acc[r];
        if (resid) v += resid[idx];
        C[idx] = v;
        if (Cbf) Cbf[idx] = f2bf(v);
    }
}

// ---------- generic WMMA GEMM, 2x2 tiles (32x32) per wave, block tile 64x64 ----------
__global__ __launch_bounds__(128) void wmma_gemm_kernel(const unsigned short* __restrict__ A,
                                                        const unsigned short* __restrict__ Bt,
                                                        float* __restrict__ C,
                                                        unsigned short* __restrict__ Cbf,
                                                        const float* __restrict__ resid,
                                                        int M, int K, int N) {
    const int lane = threadIdx.x & 31;
    const int wave = threadIdx.x >> 5;
    const int mblk = blockIdx.x * 64 + (wave >> 1) * 32;
    const int nblk = blockIdx.y * 64 + (wave & 1) * 32;
    if (mblk >= M || nblk >= N) return;          // wave-uniform

    const int sub = lane & 15;
    const int hi  = lane >> 4;

    const unsigned short* __restrict__ Ar0 = A  + (size_t)(mblk + sub) * K;
    const unsigned short* __restrict__ Ar1 = A  + (size_t)(mblk + 16 + sub) * K;
    const unsigned short* __restrict__ Br0 = Bt + (size_t)(nblk + sub) * K;
    const unsigned short* __restrict__ Br1 = Bt + (size_t)(nblk + 16 + sub) * K;

    f32x8 acc00 = {}, acc01 = {}, acc10 = {}, acc11 = {};
    for (int k = 0; k < K; k += 32) {
        FragU a0, a1, b0, b1;
#pragma unroll
        for (int j = 0; j < 8; ++j) {
            const int k0 = k + ((j >> 2) << 4) + (hi << 3) + ((j & 3) << 1);
            a0.u[j] = *(const unsigned int*)(Ar0 + k0);
            a1.u[j] = *(const unsigned int*)(Ar1 + k0);
            b0.u[j] = *(const unsigned int*)(Br0 + k0);
            b1.u[j] = *(const unsigned int*)(Br1 + k0);
        }
        acc00 = wmma_bf16(a0, b0, acc00);
        acc01 = wmma_bf16(a0, b1, acc01);
        acc10 = wmma_bf16(a1, b0, acc10);
        acc11 = wmma_bf16(a1, b1, acc11);
    }

    store_tile(acc00, mblk,      nblk,      hi, sub, C, Cbf, resid, N);
    store_tile(acc01, mblk,      nblk + 16, hi, sub, C, Cbf, resid, N);
    store_tile(acc10, mblk + 16, nblk,      hi, sub, C, Cbf, resid, N);
    store_tile(acc11, mblk + 16, nblk + 16, hi, sub, C, Cbf, resid, N);
}

// ---------- chunked parallel scan ----------
// h_s = d_s*h_{s-1} + g_s (affine, associative). 3 passes.
// Block-per-chunk: blockIdx.x = (dir*B + b)*NCH + q, threadIdx.x = channel c (DI=256).
// Wave 0 issues double-buffered TDM 2D-tile loads of o (rows of 2*DI f32) and z
// (rows of DI f32) into LDS; all waves consume after s_wait_tensorcnt + barrier.

#define SCAN_CHUNK 64
#define SCAN_SUB   8
#define NSUB       (SCAN_CHUNK / SCAN_SUB)

__device__ __forceinline__ void scan_issue_batch(const float* __restrict__ o,
                                                 const float* __restrict__ z,
                                                 int b, int L, int DI, int dir,
                                                 int s_base, int buf,
                                                 unsigned off_o, unsigned off_z) {
#if HAVE_TDM
    const int D2 = 2 * DI;
    const int l_lo = dir ? (L - s_base - SCAN_SUB) : s_base;
    const float* go = o + ((size_t)b * L + l_lo) * D2;
    const float* gz = z + ((size_t)b * L + l_lo) * DI;
    tdm_load_2d_f32(go, off_o + (unsigned)buf * SCAN_SUB * D2 * 4, D2, SCAN_SUB, D2);
    tdm_load_2d_f32(gz, off_z + (unsigned)buf * SCAN_SUB * DI * 4, DI, SCAN_SUB, DI);
#endif
}

template <int PASS>   // 1 = compose (P,S); 3 = replay with carry, emit h
__global__ __launch_bounds__(256) void scan_pass_tdm(const float* __restrict__ z,
                                                     const float* __restrict__ o_fwd,
                                                     const float* __restrict__ o_bwd,
                                                     float* __restrict__ Pbuf,
                                                     float* __restrict__ Sbuf,
                                                     const float* __restrict__ Cbuf,
                                                     unsigned short* __restrict__ hcat,
                                                     int Bb, int L, int DI) {
    extern __shared__ float lds[];
    const int NCH = L / SCAN_CHUNK;
    const int blk = blockIdx.x;
    const int q   = blk % NCH;
    const int db  = blk / NCH;            // dir*B + b
    const int b   = db % Bb;
    const int dir = db / Bb;
    const int c   = threadIdx.x;          // 0..DI-1
    const int D2  = 2 * DI;
    const float* __restrict__ o = dir ? o_bwd : o_fwd;
    const bool w0 = (threadIdx.x >> 5) == 0;   // wave-uniform

    float* o_t = lds;                                  // [2][SUB][D2]
    float* z_t = lds + 2 * SCAN_SUB * D2;              // [2][SUB][DI]
    const unsigned off_o = 0;
    const unsigned off_z = (unsigned)(2 * SCAN_SUB * D2 * 4);

    float P = 1.0f, S = 0.0f;
    float h = (PASS == 3) ? Cbuf[(size_t)blk * DI + c] : 0.0f;

#if HAVE_TDM
    if (w0) scan_issue_batch(o, z, b, L, DI, dir, q * SCAN_CHUNK, 0, off_o, off_z);
#endif
    for (int i = 0; i < NSUB; ++i) {
        const int bf = i & 1;
        const int s_base = q * SCAN_CHUNK + i * SCAN_SUB;
#if HAVE_TDM
        if (w0) {
            if (i + 1 < NSUB) {
                scan_issue_batch(o, z, b, L, DI, dir, s_base + SCAN_SUB, bf ^ 1, off_o, off_z);
                __builtin_amdgcn_s_wait_tensorcnt(2);   // previous batch complete
            } else {
                __builtin_amdgcn_s_wait_tensorcnt(0);
            }
        }
        __syncthreads();
#endif
#pragma unroll
        for (int t = 0; t < SCAN_SUB; ++t) {
            const int s = s_base + t;
            const int l = dir ? (L - 1 - s) : s;
            float o1, o2, zz;
#if HAVE_TDM
            const int r = dir ? (SCAN_SUB - 1 - t) : t;
            o1 = o_t[(bf * SCAN_SUB + r) * D2 + c];
            o2 = o_t[(bf * SCAN_SUB + r) * D2 + DI + c];
            zz = z_t[(bf * SCAN_SUB + r) * DI + c];
#else
            const size_t row = (size_t)b * L + l;
            o1 = o[row * D2 + c];
            o2 = o[row * D2 + DI + c];
            zz = z[row * DI + c];
#endif
            const float d = sigmoid_f(o1);
            if (PASS == 1) {
                S = S * d + o2 * zz;
                P = P * d;
            } else {
                h = h * d + o2 * zz;
                hcat[((size_t)b * L + l) * D2 + dir * DI + c] = f2bf(h);
            }
        }
#if HAVE_TDM
        __syncthreads();   // everyone done reading buf before it is refilled
#endif
    }
    if (PASS == 1) {
        Pbuf[(size_t)blk * DI + c] = P;
        Sbuf[(size_t)blk * DI + c] = S;
    }
}

__global__ __launch_bounds__(256) void scan_pass2(const float* __restrict__ Pbuf,
                                                  const float* __restrict__ Sbuf,
                                                  float* __restrict__ Cbuf,
                                                  int Bb, int L, int DI) {
    const int NCH = L / SCAN_CHUNK;
    const int tid = blockIdx.x * blockDim.x + threadIdx.x;
    if (tid >= 2 * Bb * DI) return;
    const int c  = tid % DI;
    const int db = tid / DI;
    float carry = 0.0f;
    for (int q = 0; q < NCH; ++q) {
        const size_t idx = ((size_t)db * NCH + q) * DI + c;
        Cbuf[idx] = carry;                 // h entering chunk q
        carry = Pbuf[idx] * carry + Sbuf[idx];
    }
}

// ---------- in-place LayerNorm over last dim (D=2048), one block per row ----------
__global__ __launch_bounds__(256) void layernorm_kernel(float* __restrict__ y,
                                                        const float* __restrict__ gamma,
                                                        const float* __restrict__ beta,
                                                        int D) {
    __shared__ float s_sum[8];
    __shared__ float s_sq[8];
    float* __restrict__ p = y + (size_t)blockIdx.x * D;

    float vals[8];
    float sum = 0.0f, sq = 0.0f;
#pragma unroll
    for (int i = 0; i < 8; ++i) {
        const float v = p[threadIdx.x + i * 256];
        vals[i] = v;
        sum += v;
        sq += v * v;
    }
#pragma unroll
    for (int off = 16; off > 0; off >>= 1) {
        sum += __shfl_down(sum, off, 32);
        sq  += __shfl_down(sq,  off, 32);
    }
    const int lane = threadIdx.x & 31;
    const int wv = threadIdx.x >> 5;
    if (lane == 0) { s_sum[wv] = sum; s_sq[wv] = sq; }
    __syncthreads();
    if (threadIdx.x == 0) {
        float ts = 0.0f, tq = 0.0f;
#pragma unroll
        for (int i = 0; i < 8; ++i) { ts += s_sum[i]; tq += s_sq[i]; }
        s_sum[0] = ts;
        s_sq[0] = tq;
    }
    __syncthreads();
    const float mean = s_sum[0] / (float)D;
    const float var  = s_sq[0] / (float)D - mean * mean;
    const float rstd = rsqrtf(var + 1e-5f);
#pragma unroll
    for (int i = 0; i < 8; ++i) {
        const int col = threadIdx.x + i * 256;
        p[col] = (vals[i] - mean) * rstd * gamma[col] + beta[col];
    }
}

extern "C" void kernel_launch(void* const* d_in, const int* in_sizes, int n_in,
                              void* d_out, int out_size, void* d_ws, size_t ws_size,
                              hipStream_t stream) {
    (void)in_sizes; (void)n_in; (void)out_size; (void)ws_size;

    const float* x     = (const float*)d_in[0];
    const float* W_in  = (const float*)d_in[1];
    const float* W_fwd = (const float*)d_in[2];
    const float* W_bwd = (const float*)d_in[3];
    const float* W_out = (const float*)d_in[4];
    const float* gamma = (const float*)d_in[5];
    const float* beta  = (const float*)d_in[6];
    float* out = (float*)d_out;

    const int Bb = 4, L = 4096, D = 2048, DI = 256;
    const int M = Bb * L;            // 16384 rows
    const int D2 = 2 * DI;           // 512
    const int NCH = L / SCAN_CHUNK;  // 64 chunks

    // ---- workspace carve-out ----
    char* w = (char*)d_ws;
    size_t off = 0;
    auto carve = [&](size_t bytes) -> void* {
        void* p = (void*)(w + off);
        off = (off + bytes + 255) & ~(size_t)255;
        return p;
    };
    unsigned short* x_bf  = (unsigned short*)carve((size_t)M * D * 2);
    unsigned short* WinT  = (unsigned short*)carve((size_t)DI * D * 2);
    unsigned short* WfwdT = (unsigned short*)carve((size_t)D2 * DI * 2);
    unsigned short* WbwdT = (unsigned short*)carve((size_t)D2 * DI * 2);
    unsigned short* WoutT = (unsigned short*)carve((size_t)D * D2 * 2);
    float*          z     = (float*)carve((size_t)M * DI * 4);
    unsigned short* z_bf  = (unsigned short*)carve((size_t)M * DI * 2);
    float*          o_fwd = (float*)carve((size_t)M * D2 * 4);
    float*          o_bwd = (float*)carve((size_t)M * D2 * 4);
    unsigned short* hcat  = (unsigned short*)carve((size_t)M * D2 * 2);
    float*          Pbuf  = (float*)carve((size_t)2 * Bb * NCH * DI * 4);
    float*          Sbuf  = (float*)carve((size_t)2 * Bb * NCH * DI * 4);
    float*          Cbuf  = (float*)carve((size_t)2 * Bb * NCH * DI * 4);

    // ---- stage conversions ----
    {
        size_t n = (size_t)M * D;
        f32_to_bf16_copy<<<(unsigned)((n + 255) / 256), 256, 0, stream>>>(x, x_bf, n);
    }
    transpose_f32_to_bf16<<<(unsigned)(((size_t)D * DI + 255) / 256), 256, 0, stream>>>(W_in,  WinT,  D,  DI);
    transpose_f32_to_bf16<<<(unsigned)(((size_t)DI * D2 + 255) / 256), 256, 0, stream>>>(W_fwd, WfwdT, DI, D2);
    transpose_f32_to_bf16<<<(unsigned)(((size_t)DI * D2 + 255) / 256), 256, 0, stream>>>(W_bwd, WbwdT, DI, D2);
    transpose_f32_to_bf16<<<(unsigned)(((size_t)D2 * D + 255) / 256), 256, 0, stream>>>(W_out, WoutT, D2, D);

    // ---- z = x @ W_in (+ bf16 copy) ----
    {
        dim3 grid(M / 64, DI / 64);
        wmma_gemm_kernel<<<grid, 128, 0, stream>>>(x_bf, WinT, z, z_bf, nullptr, M, D, DI);
    }
    // ---- o_fwd = z @ W_fwd ; o_bwd = z @ W_bwd ----
    {
        dim3 grid(M / 64, D2 / 64);
        wmma_gemm_kernel<<<grid, 128, 0, stream>>>(z_bf, WfwdT, o_fwd, nullptr, nullptr, M, DI, D2);
        wmma_gemm_kernel<<<grid, 128, 0, stream>>>(z_bf, WbwdT, o_bwd, nullptr, nullptr, M, DI, D2);
    }
    // ---- chunked parallel scans (TDM double-buffered staging) ----
    {
        const int nblocks = 2 * Bb * NCH;                       // 512 chunks
        const size_t shmem = (size_t)(2 * SCAN_SUB * D2 + 2 * SCAN_SUB * DI) * 4;  // 48KB
        scan_pass_tdm<1><<<nblocks, 256, shmem, stream>>>(z, o_fwd, o_bwd,
                                                          Pbuf, Sbuf, nullptr, nullptr, Bb, L, DI);
        const int t2 = 2 * Bb * DI;
        scan_pass2<<<(t2 + 255) / 256, 256, 0, stream>>>(Pbuf, Sbuf, Cbuf, Bb, L, DI);
        scan_pass_tdm<3><<<nblocks, 256, shmem, stream>>>(z, o_fwd, o_bwd,
                                                          nullptr, nullptr, Cbuf, hcat, Bb, L, DI);
    }
    // ---- y = hcat @ W_out + x ----
    {
        dim3 grid(M / 64, D / 64);
        wmma_gemm_kernel<<<grid, 128, 0, stream>>>(hcat, WoutT, out, nullptr, x, M, D2, D);
    }
    // ---- in-place LayerNorm ----
    layernorm_kernel<<<M, 256, 0, stream>>>(out, gamma, beta, D);
}
